// QuanvolutionNet_65481071395257
// MI455X (gfx1250) — matrix operand order repaired and checked by reading, not compiled, not run on hardware
//
#include <hip/hip_runtime.h>
#include <math.h>

typedef _Float16 v16h __attribute__((ext_vector_type(16)));
typedef _Float16 v8h  __attribute__((ext_vector_type(8)));
typedef _Float16 v4h  __attribute__((ext_vector_type(4)));
typedef float    v8f  __attribute__((ext_vector_type(8)));

#define NPATCH 196
#define KPAD   800   // 784 padded to a multiple of 32 (WMMA K) -> 25 k-steps
#define PI_F   3.14159265358979323846f

// ---------------------------------------------------------------------------
// Kernel 1: per-patch-position statistics (mean, pi/(std+eps)) over (B,4) slab
// ---------------------------------------------------------------------------
__global__ void k_stats(const float* __restrict__ x, int B,
                        float* __restrict__ stats /* [196][2] */) {
    __shared__ float sh1[256], sh2[256];
    const int p = blockIdx.x;              // 0..195
    const int r = p / 14, c = p % 14;
    const int off0 = (2 * r) * 28 + 2 * c;
    const int off1 = off0 + 28;
    float s1 = 0.f, s2 = 0.f;
    for (int b = threadIdx.x; b < B; b += blockDim.x) {
        const float* px = x + (size_t)b * 784;
        float2 a = *(const float2*)(px + off0);
        float2 d = *(const float2*)(px + off1);
        s1 += a.x + a.y + d.x + d.y;
        s2 += a.x * a.x + a.y * a.y + d.x * d.x + d.y * d.y;
    }
    sh1[threadIdx.x] = s1; sh2[threadIdx.x] = s2;
    __syncthreads();
    for (int s = 128; s > 0; s >>= 1) {
        if ((int)threadIdx.x < s) {
            sh1[threadIdx.x] += sh1[threadIdx.x + s];
            sh2[threadIdx.x] += sh2[threadIdx.x + s];
        }
        __syncthreads();
    }
    if (threadIdx.x == 0) {
        const float N = (float)B * 4.0f;
        float mean = sh1[0] / N;
        float var  = (sh2[0] - N * mean * mean) / (N - 1.0f);   // ddof=1
        var = fmaxf(var, 0.f);
        float scale = PI_F / (sqrtf(var) + 1e-6f);
        stats[p * 2 + 0] = mean;
        stats[p * 2 + 1] = scale;
    }
}

// ---------------------------------------------------------------------------
// Kernel 2: RZ-layer diagonal phases, exp(i * 0.5*sum_q (+-)params[l][q])
// diag[l][s] with s = i<<3|j<<2|k<<1|l ; bit set -> +params/2, clear -> -params/2
// ---------------------------------------------------------------------------
__global__ void k_diag(const float* __restrict__ params,
                       float* __restrict__ diag /* [2][16][2] (re,im) */) {
    const int t = threadIdx.x;             // 0..31
    if (t >= 32) return;
    const int l = t >> 4, s = t & 15;
    float ang = 0.f;
    #pragma unroll
    for (int q = 0; q < 4; ++q) {
        const int bit = (s >> (3 - q)) & 1;
        ang += (bit ? 0.5f : -0.5f) * params[l * 4 + q];
    }
    diag[t * 2 + 0] = cosf(ang);
    diag[t * 2 + 1] = sinf(ang);
}

// ---------------------------------------------------------------------------
// Kernel 3: W (10x784 f32) -> Wh (16x800 f16, B-matrix source, zero padded)
// ---------------------------------------------------------------------------
__global__ void k_wcvt(const float* __restrict__ W, _Float16* __restrict__ Wh) {
    const int i = blockIdx.x * blockDim.x + threadIdx.x;
    if (i >= 16 * KPAD) return;
    const int n = i / KPAD, k = i % KPAD;
    const float v = (n < 10 && k < 784) ? W[n * 784 + k] : 0.f;
    Wh[i] = (_Float16)v;
}

// ---------------------------------------------------------------------------
// Kernel 4: per-(b,p) 4-qubit statevector sim -> 4 f16 features
// grid: (ceil(B/256), 200); p>=196 threads just zero the K-pad columns
// ---------------------------------------------------------------------------
#define SW(a, b) { float tr_ = re[a]; re[a] = re[b]; re[b] = tr_; \
                   float ti_ = im[a]; im[a] = im[b]; im[b] = ti_; }

__global__ void k_quanv(const float* __restrict__ x, int B,
                        const float* __restrict__ stats,
                        const float* __restrict__ diag,
                        _Float16* __restrict__ feats /* [B][KPAD] */) {
    const int b = blockIdx.x * blockDim.x + threadIdx.x;
    const int p = blockIdx.y;              // 0..199 (uniform per block)
    if (b >= B) return;
    _Float16* outp = feats + (size_t)b * KPAD + p * 4;
    if (p >= NPATCH) { v4h z = {}; *(v4h*)outp = z; return; }

    const int r = p / 14, c = p % 14;
    const float mean = stats[p * 2 + 0];
    const float scl  = stats[p * 2 + 1];
    const float* px = x + (size_t)b * 784;
    float2 t0 = *(const float2*)(px + (2 * r) * 28 + 2 * c);
    float2 t1 = *(const float2*)(px + (2 * r + 1) * 28 + 2 * c);
    float pix[4] = { t0.x, t0.y, t1.x, t1.y };   // [00,01,10,11]

    float cs[4], sn[4];
    #pragma unroll
    for (int q = 0; q < 4; ++q) {
        float h = 0.5f * (pix[q] - mean) * scl;  // theta/2
        cs[q] = cosf(h);
        sn[q] = sinf(h);
    }

    // product state: real amplitudes
    float re[16], im[16];
    #pragma unroll
    for (int s = 0; s < 16; ++s) {
        float a = ((s & 8) ? sn[0] : cs[0]) * ((s & 4) ? sn[1] : cs[1]) *
                  ((s & 2) ? sn[2] : cs[2]) * ((s & 1) ? sn[3] : cs[3]);
        re[s] = a; im[s] = 0.f;
    }

    #pragma unroll
    for (int l = 0; l < 2; ++l) {
        // RZ diagonal (uniform -> scalar loads)
        #pragma unroll
        for (int s = 0; s < 16; ++s) {
            float dre = diag[(l * 16 + s) * 2 + 0];
            float dim = diag[(l * 16 + s) * 2 + 1];
            float nr = re[s] * dre - im[s] * dim;
            float ni = re[s] * dim + im[s] * dre;
            re[s] = nr; im[s] = ni;
        }
        // CNOT(0,1): control bit3, target bit2
        SW(8, 12) SW(9, 13) SW(10, 14) SW(11, 15)
        // CNOT(1,2): control bit2, target bit1
        SW(4, 6)  SW(5, 7)  SW(12, 14) SW(13, 15)
        // CNOT(2,3): control bit1, target bit0
        SW(2, 3)  SW(6, 7)  SW(10, 11) SW(14, 15)
    }

    float z0 = 0.f, z1 = 0.f, z2 = 0.f, z3 = 0.f;
    #pragma unroll
    for (int s = 0; s < 16; ++s) {
        float pr = re[s] * re[s] + im[s] * im[s];
        z0 += (s & 8) ? -pr : pr;
        z1 += (s & 4) ? -pr : pr;
        z2 += (s & 2) ? -pr : pr;
        z3 += (s & 1) ? -pr : pr;
    }
    v4h o; o[0] = (_Float16)z0; o[1] = (_Float16)z1;
           o[2] = (_Float16)z2; o[3] = (_Float16)z3;
    *(v4h*)outp = o;
}

// ---------------------------------------------------------------------------
// Kernel 5: WMMA GEMM  C[B][16] = feats[B][800] @ Wh[16][800]^T
// one wave32 per 16-row tile; 25 x v_wmma_f32_16x16x32_f16
// ---------------------------------------------------------------------------
union U16h { v16h v; v8h h[2]; };

__global__ void k_gemm(const _Float16* __restrict__ A  /* [B][KPAD] */,
                       const _Float16* __restrict__ Bh /* [16][KPAD] */,
                       float* __restrict__ Craw        /* [B][16] */,
                       int Brows) {
    const int wave = (blockIdx.x * blockDim.x + threadIdx.x) >> 5;  // tile id
    const int lane = threadIdx.x & 31;
    const int lo = lane & 15, hi = lane >> 4;
    const int row0 = wave * 16;
    if (row0 >= Brows) return;   // wave-uniform; EXEC all-ones inside

    // A: lane lo (hi=0): row row0+lo, K {k0..k0+7, k0+16..k0+23}
    //    lane lo (hi=1): row row0+lo, K {k0+8..+15, k0+24..+31}
    const _Float16* aBase = A + (size_t)(row0 + lo) * KPAD + hi * 8;
    // B: lane lo holds column n=lo; hi=0 -> K {k0..k0+7, k0+16..k0+23}, hi=1 -> K {k0+8..+15, k0+24..+31}
    const _Float16* bBase = Bh + (size_t)lo * KPAD + hi * 8;

    v8f acc = {};
    for (int k0 = 0; k0 < KPAD; k0 += 32) {
        U16h a, bb;
        a.h[0]  = *(const v8h*)(aBase + k0);
        a.h[1]  = *(const v8h*)(aBase + k0 + 16);
        bb.h[0] = *(const v8h*)(bBase + k0);
        bb.h[1] = *(const v8h*)(bBase + k0 + 16);
        acc = __builtin_amdgcn_wmma_f32_16x16x32_f16(
            /*neg_a=*/false, a.v, /*neg_b=*/false, bb.v,
            /*c_mod=*/(short)0, acc, /*reuse_a=*/false, /*reuse_b=*/false);
    }

    // C/D layout: VGPR j -> (M=j, N=lane) lanes 0-15 ; (M=8+j, N=lane-16) lanes 16-31
    #pragma unroll
    for (int j = 0; j < 8; ++j) {
        const int row = row0 + j + hi * 8;
        Craw[(size_t)row * 16 + lo] = acc[j];
    }
}

// ---------------------------------------------------------------------------
// Kernel 6: bias + log_softmax over 10 classes
// ---------------------------------------------------------------------------
__global__ void k_lsm(const float* __restrict__ Craw, const float* __restrict__ bias,
                      float* __restrict__ out, int B) {
    const int b = blockIdx.x * blockDim.x + threadIdx.x;
    if (b >= B) return;
    float v[10];
    float mx = -INFINITY;
    #pragma unroll
    for (int n = 0; n < 10; ++n) {
        v[n] = Craw[(size_t)b * 16 + n] + bias[n];
        mx = fmaxf(mx, v[n]);
    }
    float se = 0.f;
    #pragma unroll
    for (int n = 0; n < 10; ++n) se += expf(v[n] - mx);
    const float lse = logf(se) + mx;
    #pragma unroll
    for (int n = 0; n < 10; ++n) out[(size_t)b * 10 + n] = v[n] - lse;
}

// ---------------------------------------------------------------------------
extern "C" void kernel_launch(void* const* d_in, const int* in_sizes, int n_in,
                              void* d_out, int out_size, void* d_ws, size_t ws_size,
                              hipStream_t stream) {
    const float* x      = (const float*)d_in[0];  // (B,28,28)
    const float* params = (const float*)d_in[1];  // (2,4)
    const float* W      = (const float*)d_in[2];  // (10,784)
    const float* bias   = (const float*)d_in[3];  // (10,)
    float* out = (float*)d_out;
    const int B = in_sizes[0] / 784;              // 4096

    // workspace layout (16B-aligned slices)
    char* ws = (char*)d_ws;
    float*    stats = (float*)ws;                           // 196*2 f32
    float*    diag  = stats + 2 * NPATCH;                   // 2*16*2 f32
    _Float16* Wh    = (_Float16*)(diag + 64);               // 16*800 f16
    _Float16* feats = Wh + 16 * KPAD;                       // B*800 f16
    float*    Craw  = (float*)(feats + (size_t)B * KPAD);   // B*16 f32

    k_stats<<<NPATCH, 256, 0, stream>>>(x, B, stats);
    k_diag <<<1, 32, 0, stream>>>(params, diag);
    k_wcvt <<<(16 * KPAD + 255) / 256, 256, 0, stream>>>(W, Wh);

    dim3 g2((B + 255) / 256, 200);   // y = 196 patches + 4 pad-column blocks
    k_quanv<<<g2, 256, 0, stream>>>(x, B, stats, diag, feats);

    const int tiles = B / 16;                     // 256 waves
    k_gemm<<<(tiles + 7) / 8, 256, 0, stream>>>(feats, Wh, Craw, B);

    k_lsm<<<(B + 255) / 256, 256, 0, stream>>>(Craw, bias, out, B);
}
